// PiKVLLM_45999099740748
// MI455X (gfx1250) — matrix operand order repaired and checked by reading, not compiled, not run on hardware
//
#include <hip/hip_runtime.h>
#include <hip/hip_bf16.h>
#include <math.h>

// ---------------- model dims (must match reference) ----------------
#define V_SZ    32000
#define C_DIM   1024
#define H_HEADS 16
#define D_HEAD  64
#define E_EXP   8
#define L_LAYERS 4
#define B_SZ    4
#define T_SEQ   512
#define BT      (B_SZ * T_SEQ)          // 2048 tokens
#define KV_BASE 128
#define KV_DEC  8

typedef __bf16 bf16_t;
typedef __bf16 bf16x16 __attribute__((ext_vector_type(16)));
typedef float  f32x8   __attribute__((ext_vector_type(8)));

// single place for fp32 -> bf16; compiler emits the v_cvt hardware path
__device__ __forceinline__ bf16_t f2bf(float f) { return (bf16_t)f; }

struct Frag2x16B { uint4 lo, hi; };

// Assemble a 16-element bf16 WMMA fragment from two contiguous 16B runs.
__device__ __forceinline__ bf16x16 load_frag(const bf16_t* p0, const bf16_t* p1) {
  Frag2x16B t;
  t.lo = *(const uint4*)p0;
  t.hi = *(const uint4*)p1;
  return __builtin_bit_cast(bf16x16, t);
}

// =====================================================================
// GEMM: Cout[M=2048, N] = A[M,K]bf16 @ W[N,K]bf16^T (+bias)(+GELU)
// 256-thread block = 8 waves. Block tile 64(M) x 128(N).
// Double-buffered LDS software pipeline, one barrier per 32-wide K chunk:
//   iter k: global_load chunk k+1 -> regs ; wmma on LDS[cur] ;
//           regs -> LDS[nxt] ; barrier
// Wave w: rows [(w&3)*16 .. +16), cols [(w>>2)*64 .. +64) of block tile
// => A-fragment reused 4x; 4 wmma/wave/chunk, 32 wmma/block/chunk.
// All four B fragments preloaded before the wmma chain (single dscnt wait).
//
// Fragment layouts used here (wave32), bf16 16x16x32, k = half*16 + elem:
//   A: lane m=lane&15, half=lane>>4; elems 0..7 = A[m][half*16 .. +7],
//      elems 8..15 = A[m][half*16+8 .. +7]  -> two 16B LDS reads
//   B: lane n=lane&15; elems 0..15 = B[k=half*16 .. +15][n]
//      (stored here as W[n][k], k contiguous) -> two 16B LDS reads
//   C/D: vgpr v -> row = v + half*8, col = lane&15
// =====================================================================
__global__ __launch_bounds__(256) void gemm_bf16_wmma(
    const bf16_t* __restrict__ A, const bf16_t* __restrict__ W,
    const float* __restrict__ bias, void* __restrict__ Cout,
    int N, int K, int gelu, int out_bf16) {
  __shared__ __align__(16) bf16_t As[2][64 * 32];
  __shared__ __align__(16) bf16_t Bs[2][128 * 32];

  const int tid   = threadIdx.x;
  const int lane  = tid & 31;
  const int wave  = tid >> 5;
  const int half  = lane >> 4;
  const int l15   = lane & 15;
  const int mBase = blockIdx.y * 64;
  const int nBase = blockIdx.x * 128;
  const int mSub  = (wave & 3) * 16;        // row sub-tile within block
  const int nOff  = (wave >> 2) * 64;       // col half within block

  const int srow = tid >> 2;                // staging row 0..63
  const int squad = (tid & 3) * 8;          // staging 8-elem quad offset

  const bf16_t* aSrcBase  = A + (size_t)(mBase + srow) * K + squad;
  const bf16_t* bSrcBase0 = W + (size_t)(nBase + srow) * K + squad;
  const bf16_t* bSrcBase1 = W + (size_t)(nBase + 64 + srow) * K + squad;
  bf16_t* aDst0 = &As[0][srow * 32 + squad];
  bf16_t* bDst0 = &Bs[0][srow * 32 + squad];
  bf16_t* bDst1 = &Bs[0][(64 + srow) * 32 + squad];
  const size_t bufStrideA = 64 * 32;        // elems between As[0] and As[1]
  const size_t bufStrideB = 128 * 32;

  f32x8 acc[4];
  #pragma unroll
  for (int s = 0; s < 4; ++s)
    acc[s] = (f32x8){0.f, 0.f, 0.f, 0.f, 0.f, 0.f, 0.f, 0.f};

  auto compute = [&](int cur) {
    const bf16_t* ap = &As[cur][(mSub + l15) * 32];
    bf16x16 a = load_frag(ap + half * 16, ap + half * 16 + 8);
    bf16x16 bf[4];
    #pragma unroll
    for (int s = 0; s < 4; ++s) {
      const bf16_t* bp = &Bs[cur][(nOff + s * 16 + l15) * 32 + half * 16];
      bf[s] = load_frag(bp, bp + 8);
    }
    #pragma unroll
    for (int s = 0; s < 4; ++s)
      acc[s] = __builtin_amdgcn_wmma_f32_16x16x32_bf16(
          false, a, false, bf[s], (short)0, acc[s], false, false);
  };

  // ---- prologue: stage chunk 0 into buffer 0 ----
  {
    uint4 rA  = *(const uint4*)aSrcBase;
    uint4 rB0 = *(const uint4*)bSrcBase0;
    uint4 rB1 = *(const uint4*)bSrcBase1;
    *(uint4*)aDst0 = rA;
    *(uint4*)bDst0 = rB0;
    *(uint4*)bDst1 = rB1;
  }
  __syncthreads();

  const int K32 = K >> 5;
  for (int kc = 0; kc + 1 < K32; ++kc) {
    const int cur = kc & 1, nxt = cur ^ 1;
    const int kn = (kc + 1) << 5;
    // global loads for next chunk (in flight across the wmma chain)
    uint4 rA  = *(const uint4*)(aSrcBase  + kn);
    uint4 rB0 = *(const uint4*)(bSrcBase0 + kn);
    uint4 rB1 = *(const uint4*)(bSrcBase1 + kn);
    // speculative prefetch of chunk kc+2 (safe past end on gfx1250)
    __builtin_prefetch(aSrcBase  + kn + 32, 0, 1);
    __builtin_prefetch(bSrcBase0 + kn + 32, 0, 1);
    __builtin_prefetch(bSrcBase1 + kn + 32, 0, 1);

    compute(cur);

    *(uint4*)(aDst0 + (size_t)nxt * bufStrideA) = rA;
    *(uint4*)(bDst0 + (size_t)nxt * bufStrideB) = rB0;
    *(uint4*)(bDst1 + (size_t)nxt * bufStrideB) = rB1;
    __syncthreads();
  }
  compute((K32 - 1) & 1);

  // ---- epilogue: bias (+GELU), store f32 or bf16 ----
  #pragma unroll
  for (int s = 0; s < 4; ++s) {
    int col = nBase + nOff + s * 16 + l15;
    float bv = bias ? bias[col] : 0.f;
    #pragma unroll
    for (int v = 0; v < 8; ++v) {
      int row = mBase + mSub + v + (half << 3);
      float val = acc[s][v] + bv;
      if (gelu) val = 0.5f * val * (1.0f + erff(val * 0.70710678118654752f));
      if (out_bf16) ((bf16_t*)Cout)[(size_t)row * N + col] = f2bf(val);
      else          ((float*)Cout)[(size_t)row * N + col] = val;
    }
  }
}

// =====================================================================
// Streaming fp32 -> bf16 (weights, per use). n must be a multiple of 4.
// =====================================================================
__global__ __launch_bounds__(256) void f32_to_bf16_kernel(
    const float* __restrict__ in, bf16_t* __restrict__ out, long n) {
  long i = ((long)blockIdx.x * 256 + threadIdx.x) * 4;
  if (i < n) {
    float4 f = *(const float4*)(in + i);
    out[i + 0] = f2bf(f.x);
    out[i + 1] = f2bf(f.y);
    out[i + 2] = f2bf(f.z);
    out[i + 3] = f2bf(f.w);
  }
}

// =====================================================================
// Embedding gather + sinusoidal PE; emits f32 x and bf16 xbf.
// =====================================================================
__global__ __launch_bounds__(256) void embed_pe_kernel(
    const int* __restrict__ ids, const float* __restrict__ emb,
    float* __restrict__ x, bf16_t* __restrict__ xbf) {
  int idx = blockIdx.x * 256 + threadIdx.x;     // over BT*C
  int c  = idx & (C_DIM - 1);
  int bt = idx >> 10;                           // C_DIM == 1024
  int t  = bt & (T_SEQ - 1);                    // T_SEQ == 512
  int id = ids[bt];
  int i2 = c & ~1;
  float freq = __expf(-(float)i2 * (9.210340371976184f / (float)C_DIM)); // ln(1e4)
  float ang  = (float)t * freq;
  float pe   = (c & 1) ? __cosf(ang) : __sinf(ang);
  float v = emb[(size_t)id * C_DIM + c] + pe;
  x[idx]   = v;
  xbf[idx] = f2bf(v);
}

// =====================================================================
// Flash-style attention: wave32 per (b,h,t); lane holds 2 of D=64.
// Writes bf16 directly (consumed only by the out-proj WMMA GEMM).
// =====================================================================
__global__ __launch_bounds__(256) void attn_kernel(
    const float* __restrict__ qkv, bf16_t* __restrict__ out) {
  int lane = threadIdx.x & 31;
  int wid  = blockIdx.x * 8 + (threadIdx.x >> 5);    // B*H*T waves
  int t  = wid % T_SEQ;
  int bh = wid / T_SEQ;
  int h  = bh % H_HEADS;
  int b  = bh / H_HEADS;

  const float* qp = qkv + ((size_t)(b * T_SEQ + t)) * (3 * C_DIM) + h * D_HEAD;
  float q0 = qp[2 * lane], q1 = qp[2 * lane + 1];

  float m = -1e30f, l = 0.f, o0 = 0.f, o1 = 0.f;
  for (int s = 0; s < T_SEQ; ++s) {
    const float* kp = qkv + ((size_t)(b * T_SEQ + s)) * (3 * C_DIM) + C_DIM + h * D_HEAD;
    float part = q0 * kp[2 * lane] + q1 * kp[2 * lane + 1];
    #pragma unroll
    for (int off = 16; off > 0; off >>= 1) part += __shfl_xor(part, off, 32);
    float score = part * 0.125f;                     // 1/sqrt(64)
    float mn   = fmaxf(m, score);
    float corr = __expf(m - mn);
    float p    = __expf(score - mn);
    const float* vp = qkv + ((size_t)(b * T_SEQ + s)) * (3 * C_DIM) + 2 * C_DIM + h * D_HEAD;
    l  = l * corr + p;
    o0 = o0 * corr + p * vp[2 * lane];
    o1 = o1 * corr + p * vp[2 * lane + 1];
    m = mn;
  }
  float inv = 1.0f / l;
  bf16_t* op = out + ((size_t)(b * T_SEQ + t)) * C_DIM + h * D_HEAD;
  op[2 * lane]     = f2bf(o0 * inv);
  op[2 * lane + 1] = f2bf(o1 * inv);
}

// =====================================================================
// out = LayerNorm(x + r)*g + b   (r nullable). Also emits bf16 copy.
// One block per token row.
// =====================================================================
__global__ __launch_bounds__(256) void add_ln_kernel(
    const float* __restrict__ x, const float* __restrict__ r,
    const float* __restrict__ g, const float* __restrict__ b,
    float* __restrict__ out, bf16_t* __restrict__ out_bf) {
  __shared__ float red[256];
  __shared__ float s_stat;
  int t = blockIdx.x;
  float loc[4];
  float sum = 0.f;
  #pragma unroll
  for (int j = 0; j < 4; ++j) {
    int c = j * 256 + threadIdx.x;
    float v = x[(size_t)t * C_DIM + c];
    if (r) v += r[(size_t)t * C_DIM + c];
    loc[j] = v;
    sum += v;
  }
  red[threadIdx.x] = sum; __syncthreads();
  for (int s = 128; s > 0; s >>= 1) {
    if (threadIdx.x < s) red[threadIdx.x] += red[threadIdx.x + s];
    __syncthreads();
  }
  if (threadIdx.x == 0) s_stat = red[0] * (1.0f / C_DIM);
  __syncthreads();
  float mean = s_stat;
  float vs = 0.f;
  #pragma unroll
  for (int j = 0; j < 4; ++j) { float d = loc[j] - mean; vs += d * d; }
  __syncthreads();
  red[threadIdx.x] = vs; __syncthreads();
  for (int s = 128; s > 0; s >>= 1) {
    if (threadIdx.x < s) red[threadIdx.x] += red[threadIdx.x + s];
    __syncthreads();
  }
  if (threadIdx.x == 0) s_stat = rsqrtf(red[0] * (1.0f / C_DIM) + 1e-5f);
  __syncthreads();
  float rstd = s_stat;
  #pragma unroll
  for (int j = 0; j < 4; ++j) {
    int c = j * 256 + threadIdx.x;
    float o = (loc[j] - mean) * rstd * g[c] + b[c];
    out[(size_t)t * C_DIM + c] = o;
    if (out_bf) out_bf[(size_t)t * C_DIM + c] = f2bf(o);
  }
}

// =====================================================================
// Router: wave32 per token; top-2 of logits (softmax monotonic);
// winner = max(top-2 indices) == reference "last scatter write wins".
// =====================================================================
__global__ __launch_bounds__(256) void router_kernel(
    const float* __restrict__ x, const float* __restrict__ rw,
    const float* __restrict__ rb, float* __restrict__ maskf,
    int* __restrict__ winner) {
  int lane = threadIdx.x & 31;
  int t = blockIdx.x * 8 + (threadIdx.x >> 5);
  float acc[E_EXP];
  #pragma unroll
  for (int e = 0; e < E_EXP; ++e) acc[e] = 0.f;
  for (int c = lane; c < C_DIM; c += 32) {
    float xv = x[(size_t)t * C_DIM + c];
    #pragma unroll
    for (int e = 0; e < E_EXP; ++e) acc[e] += xv * rw[e * C_DIM + c];
  }
  #pragma unroll
  for (int e = 0; e < E_EXP; ++e) {
    #pragma unroll
    for (int off = 16; off > 0; off >>= 1) acc[e] += __shfl_xor(acc[e], off, 32);
  }
  if (lane == 0) {
    #pragma unroll
    for (int e = 0; e < E_EXP; ++e) acc[e] += rb[e];
    int i1 = 0;
    for (int e = 1; e < E_EXP; ++e) if (acc[e] > acc[i1]) i1 = e;   // ties -> lowest
    int i2 = (i1 == 0) ? 1 : 0;
    for (int e = 0; e < E_EXP; ++e) if (e != i1 && acc[e] > acc[i2]) i2 = e;
    winner[t] = (i1 > i2) ? i1 : i2;
    for (int e = 0; e < E_EXP; ++e)
      maskf[t * E_EXP + e] = (e == i1 || e == i2) ? 1.f : 0.f;
  }
}

// cached[e][c] = (sum_t y[t][c]*mask[t][e]) / max(count_e,1) / cache_size
__global__ __launch_bounds__(256) void moe_mean_kernel(
    const float* __restrict__ y, const float* __restrict__ maskf,
    float* __restrict__ cached, int e, float inv_cache) {
  int c = blockIdx.x * 256 + threadIdx.x;
  float s = 0.f, cnt = 0.f;
  for (int t = 0; t < BT; ++t) {
    float mk = maskf[t * E_EXP + e];
    s   += y[(size_t)t * C_DIM + c] * mk;
    cnt += mk;
  }
  cached[e * C_DIM + c] = s / fmaxf(cnt, 1.0f) * inv_cache;
}

// out = (init?0:out) + (winner[t]==e ? y : 0)
__global__ __launch_bounds__(256) void moe_select_kernel(
    const float* __restrict__ y, const int* __restrict__ winner,
    float* __restrict__ out, int e, int init) {
  int idx = blockIdx.x * 256 + threadIdx.x;
  int t = idx >> 10;                         // / C_DIM
  float v = (winner[t] == e) ? y[idx] : 0.f;
  out[idx] = init ? v : (out[idx] + v);
}

// out[t][c] += cached[winner[t]][c]
__global__ __launch_bounds__(256) void moe_cache_kernel(
    float* __restrict__ out, const float* __restrict__ cached,
    const int* __restrict__ winner) {
  int idx = blockIdx.x * 256 + threadIdx.x;
  int t = idx >> 10;
  int c = idx & (C_DIM - 1);
  out[idx] += cached[winner[t] * C_DIM + c];
}

// =====================================================================
extern "C" void kernel_launch(void* const* d_in, const int* in_sizes, int n_in,
                              void* d_out, int out_size, void* d_ws, size_t ws_size,
                              hipStream_t stream) {
  (void)in_sizes; (void)n_in; (void)out_size; (void)ws_size;

  const int*   ids    = (const int*)  d_in[0];
  const float* emb    = (const float*)d_in[1];
  const float* qkv_w  = (const float*)d_in[2];
  const float* qkv_b  = (const float*)d_in[3];
  const float* out_w  = (const float*)d_in[4];
  const float* out_b  = (const float*)d_in[5];
  const float* ln1_g  = (const float*)d_in[6];
  const float* ln1_b  = (const float*)d_in[7];
  const float* rw     = (const float*)d_in[8];
  const float* rb     = (const float*)d_in[9];
  const float* w1     = (const float*)d_in[10];
  const float* b1     = (const float*)d_in[11];
  const float* w2     = (const float*)d_in[12];
  const float* b2     = (const float*)d_in[13];
  const float* ln2_g  = (const float*)d_in[14];
  const float* ln2_b  = (const float*)d_in[15];
  const float* lnf_g  = (const float*)d_in[16];
  const float* lnf_b  = (const float*)d_in[17];
  const float* head_w = (const float*)d_in[18];

  // ---- workspace layout; ~134 MB ----
  const size_t BTC = (size_t)BT * C_DIM;
  float*  ws     = (float*)d_ws;
  float*  x      = ws;                                  // [BT,C] f32
  float*  qkv    = x      + BTC;                        // [BT,3C] f32
  float*  proj   = qkv    + 3 * BTC;                    // [BT,C] f32
  float*  ybuf   = proj   + BTC;                        // [BT,C] f32
  float*  moeout = ybuf   + BTC;                        // [BT,C] f32
  float*  cached = moeout + BTC;                        // [E,C] f32
  float*  maskf  = cached + (size_t)E_EXP * C_DIM;      // [BT,E] f32
  int*    winner = (int*)(maskf + (size_t)BT * E_EXP);  // [BT]
  bf16_t* xbf    = (bf16_t*)(winner + BT);              // [BT,C] bf16
  bf16_t* attnbf = xbf    + BTC;                        // [BT,C] bf16
  bf16_t* hbf    = attnbf + BTC;                        // [BT,C] bf16
  bf16_t* wbf    = hbf    + BTC;                        // up to [V,C] bf16 (64 MB)

  dim3 b256(256);
  const unsigned gM = BT / 64;                          // 32 row-blocks

  auto cvt = [&](const float* src, long n) {
    f32_to_bf16_kernel<<<dim3((unsigned)((n / 4 + 255) / 256)), b256, 0, stream>>>(
        src, wbf, n);
  };

  embed_pe_kernel<<<dim3((unsigned)(BTC / 256)), b256, 0, stream>>>(ids, emb, x, xbf);

  for (int l = 0; l < L_LAYERS; ++l) {
    // ---- QKV projection: [BT,C] x [3C,C]^T -> qkv f32 ----
    cvt(qkv_w + (size_t)l * 3 * C_DIM * C_DIM, (long)3 * C_DIM * C_DIM);
    gemm_bf16_wmma<<<dim3(3 * C_DIM / 128, gM), b256, 0, stream>>>(
        xbf, wbf, qkv_b + (size_t)l * 3 * C_DIM, qkv, 3 * C_DIM, C_DIM, 0, 0);

    attn_kernel<<<dim3(B_SZ * H_HEADS * T_SEQ / 8), b256, 0, stream>>>(qkv, attnbf);

    // ---- output projection ----
    cvt(out_w + (size_t)l * C_DIM * C_DIM, (long)C_DIM * C_DIM);
    gemm_bf16_wmma<<<dim3(C_DIM / 128, gM), b256, 0, stream>>>(
        attnbf, wbf, out_b + (size_t)l * C_DIM, proj, C_DIM, C_DIM, 0, 0);

    add_ln_kernel<<<dim3(BT), b256, 0, stream>>>(
        x, proj, ln1_g + (size_t)l * C_DIM, ln1_b + (size_t)l * C_DIM, x, xbf);

    router_kernel<<<dim3(BT / 8), b256, 0, stream>>>(
        x, rw + (size_t)l * E_EXP * C_DIM, rb + (size_t)l * E_EXP, maskf, winner);

    const float inv_cache = 1.0f / (float)(KV_BASE - l * KV_DEC);
    for (int e = 0; e < E_EXP; ++e) {
      const float* w1e = w1 + ((size_t)l * E_EXP + e) * C_DIM * C_DIM;
      const float* b1e = b1 + ((size_t)l * E_EXP + e) * C_DIM;
      const float* w2e = w2 + ((size_t)l * E_EXP + e) * C_DIM * C_DIM;
      const float* b2e = b2 + ((size_t)l * E_EXP + e) * C_DIM;
      // h = gelu(x@w1^T+b1) emitted directly as bf16 (only feeds gemm2)
      cvt(w1e, (long)C_DIM * C_DIM);
      gemm_bf16_wmma<<<dim3(C_DIM / 128, gM), b256, 0, stream>>>(
          xbf, wbf, b1e, hbf, C_DIM, C_DIM, /*gelu=*/1, /*out_bf16=*/1);
      cvt(w2e, (long)C_DIM * C_DIM);
      gemm_bf16_wmma<<<dim3(C_DIM / 128, gM), b256, 0, stream>>>(
          hbf, wbf, b2e, ybuf, C_DIM, C_DIM, 0, 0);
      moe_select_kernel<<<dim3((unsigned)(BTC / 256)), b256, 0, stream>>>(
          ybuf, winner, moeout, e, e == 0);
      moe_mean_kernel<<<dim3(C_DIM / 256), b256, 0, stream>>>(
          ybuf, maskf, cached, e, inv_cache);
    }
    moe_cache_kernel<<<dim3((unsigned)(BTC / 256)), b256, 0, stream>>>(
        moeout, cached, winner);

    add_ln_kernel<<<dim3(BT), b256, 0, stream>>>(
        x, moeout, ln2_g + (size_t)l * C_DIM, ln2_b + (size_t)l * C_DIM, x, xbf);
  }

  add_ln_kernel<<<dim3(BT), b256, 0, stream>>>(x, nullptr, lnf_g, lnf_b, x, xbf);

  // ---- LM head: [BT,C] x [V,C]^T -> d_out f32, no bias ----
  cvt(head_w, (long)V_SZ * C_DIM);
  gemm_bf16_wmma<<<dim3(V_SZ / 128, gM), b256, 0, stream>>>(
      xbf, wbf, nullptr, d_out, V_SZ, C_DIM, 0, 0);
}
